// DilatedAttention_75411035783274
// MI455X (gfx1250) — compile-verified, matrix operand order checked by reading, not run on hardware
//
#include <hip/hip_runtime.h>

// ---- problem constants -----------------------------------------------------
// B=4, S=4096, D=768, H=12, HD=64, DIL=8 -> Sd=512
#define CB   4
#define CS   4096
#define CD   768
#define CH   12
#define CHD  64
#define CSD  512

typedef __attribute__((ext_vector_type(16))) _Float16 v16h;
typedef __attribute__((ext_vector_type(8)))  _Float16 v8h;
typedef __attribute__((ext_vector_type(8)))  float    v8f;
typedef __attribute__((ext_vector_type(4)))  float    v4f;

// Per-lane 16-half WMMA A/B fragment from two contiguous 16B chunks
// (16-bit 16x32 operand: lanes 0-15 K={0..7,16..23}, lanes 16-31 K={8..15,24..31};
//  caller pre-offsets p by (lane>=16 ? 8 : 0)).
__device__ __forceinline__ v16h ld_frag(const _Float16* p) {
    v8h lo = *(const v8h*)(p);
    v8h hi = *(const v8h*)(p + 16);
    return __builtin_shufflevector(lo, hi, 0,1,2,3,4,5,6,7,8,9,10,11,12,13,14,15);
}

__device__ __forceinline__ v8f wmma16(v16h a, v16h b, v8f c) {
    return __builtin_amdgcn_wmma_f32_16x16x32_f16(false, a, false, b, (short)0, c,
                                                  false, false);
}

// ---- CDNA5 async global->LDS copy (ASYNCcnt), 16B per lane -----------------
__device__ __forceinline__ void async_cp16(uint32_t lds_addr, const void* base,
                                           uint32_t byte_off) {
    asm volatile("global_load_async_to_lds_b128 %0, %1, %2"
                 :: "v"(lds_addr), "v"(byte_off), "s"(base) : "memory");
}
__device__ __forceinline__ void wait_async0() {
    asm volatile("s_wait_asynccnt 0" ::: "memory");
}
__device__ __forceinline__ uint32_t lds_off(const void* p) {
    // LDS aperture: generic address low 32 bits are the DS byte offset
    return (uint32_t)(uintptr_t)p;
}

// ---- f32 -> f16 cast -------------------------------------------------------
__global__ __launch_bounds__(256)
void cast_f32_f16(const float* __restrict__ in, _Float16* __restrict__ out, int n) {
    int i = blockIdx.x * blockDim.x + threadIdx.x;
    int stride = gridDim.x * blockDim.x;
    for (; i < n; i += stride) out[i] = (_Float16)in[i];
}

// ---- WMMA GEMM: C[M,768] = A[M,768] @ W[768,768]^T + bias ------------------
// Block: 8 waves -> 128 rows x 128 cols. W tile (128 x 64 halfs) staged in LDS
// via async copies, double-buffered; each wave computes 8 16x16 tiles.
// MODE 0: Q (f16 natural)   MODE 1: K_d (gathered rows, f16 natural)
// MODE 2: V_d (gathered rows, f16 transposed to (b,h,hd,key))
// MODE 3: output projection (f32 natural)
template <int MODE>
__global__ __launch_bounds__(256)
void gemm_wmma(const _Float16* __restrict__ A, const _Float16* __restrict__ W,
               const float* __restrict__ bias, _Float16* __restrict__ outh,
               float* __restrict__ outf) {
    __shared__ _Float16 sW[2][128 * 64];   // 2 x 16KB

    const int tid  = threadIdx.x;
    const int lane = tid & 31, wave = tid >> 5;
    const int l16  = lane & 15, hig = lane >> 4, ks = hig * 8;
    const int m0   = (blockIdx.y * 8 + wave) * 16;
    const int n0   = blockIdx.x * 128;

    int arow = m0 + l16;
    if (MODE == 1 || MODE == 2)            // gather dilated rows of x
        arow = (arow >> 9) * CS + (arow & (CSD - 1)) * 8;
    const _Float16* Arow = A + (size_t)arow * CD;

    // cooperative async W-tile load: 1024 16B pieces, 4 per thread
    auto load_w = [&](int kc, int buf) {
#pragma unroll
        for (int i = 0; i < 4; ++i) {
            const int p   = tid * 4 + i;
            const int row = p >> 3, kof = (p & 7) * 8;
            async_cp16(lds_off(&sW[buf][0]) + (uint32_t)p * 16, W,
                       (uint32_t)(((n0 + row) * CD + kc + kof) * 2));
        }
    };

    v8f acc[8] = {};
    load_w(0, 0);
    for (int c = 0; c < CD / 64; ++c) {
        wait_async0();
        __syncthreads();
        if (c < CD / 64 - 1) load_w((c + 1) * 64, (c + 1) & 1);
        const _Float16* wb = &sW[c & 1][0];
        __builtin_prefetch(Arow + c * 64 + 64, 0, 1);
#pragma unroll
        for (int kh = 0; kh < 2; ++kh) {
            v16h a = ld_frag(Arow + c * 64 + kh * 32 + ks);
            // hoist all 8 B fragments -> one LDS load clause, one wait
            v16h bfr[8];
#pragma unroll
            for (int t = 0; t < 8; ++t)
                bfr[t] = ld_frag(wb + (t * 16 + l16) * 64 + kh * 32 + ks);
#pragma unroll
            for (int t = 0; t < 8; ++t)
                acc[t] = wmma16(a, bfr[t], acc[t]);
        }
    }

#pragma unroll
    for (int t = 0; t < 8; ++t) {
        const int ncol = n0 + t * 16 + l16;
        const float bv = bias[ncol];
#pragma unroll
        for (int r = 0; r < 8; ++r) {
            const int row = m0 + r + 8 * hig;
            const float v = acc[t][r] + bv;
            if (MODE == 3) {
                outf[(size_t)row * CD + ncol] = v;
            } else if (MODE == 2) {
                const int bidx = row >> 9, key = row & (CSD - 1);
                const int hh = ncol >> 6, d = ncol & 63;
                outh[(((size_t)(bidx * CH + hh) * CHD + d) << 9) + key] = (_Float16)v;
            } else {
                outh[(size_t)row * CD + ncol] = (_Float16)v;
            }
        }
    }
}

// ---- flash attention over dilated keys ------------------------------------
// One block per (b, h, 128-query group); 8 waves each own 16 queries.
// K/V chunks (32 keys) staged in LDS with async copies, double-buffered.
// Softmax runs in a transposed domain (2 cross-lane ops per chunk).
__global__ __launch_bounds__(256)
void attn_wmma(const _Float16* __restrict__ q, const _Float16* __restrict__ kd,
               const _Float16* __restrict__ vdt, _Float16* __restrict__ ctx) {
    __shared__ _Float16 sK[2][32 * 64];    // keys x hd          (8KB)
    __shared__ _Float16 sV[2][64 * 32];    // hd x keys          (8KB)
    __shared__ float    sS[8][16 * 32];    // per-wave scores    (16KB)
    __shared__ _Float16 sP[8][16 * 32];    // per-wave P tile    (8KB)
    __shared__ float    sA[8][16];         // per-wave row alpha
    __shared__ float    sL[8][16];         // per-wave row l

    const int tid  = threadIdx.x;
    const int lane = tid & 31, wave = tid >> 5;
    const int l16  = lane & 15, hig = lane >> 4, ks = hig * 8;

    const int blk = blockIdx.x;            // B*H*32 = 1536
    const int b   = blk / (CH * 32);
    const int rem = blk % (CH * 32);
    const int h   = rem / 32;
    const int qg  = rem % 32;

    const int qrow0 = b * CS + (qg * 8 + wave) * 16;
    const _Float16* qbase = q   + (size_t)qrow0 * CD + h * CHD;
    const _Float16* kbase = kd  + (size_t)(b * CSD) * CD + h * CHD;   // block-uniform
    const _Float16* vbase = vdt + ((size_t)(b * CH + h) * CHD) * CSD; // block-uniform

    // stage one 32-key chunk of K (32x64 halfs) and V^T (64x32 halfs): 16B/thread each
    auto load_kv = [&](int kk, int buf) {
        { const int p = tid, row = p >> 3, kof = (p & 7) * 8;
          async_cp16(lds_off(&sK[buf][0]) + (uint32_t)p * 16, kbase,
                     (uint32_t)(((kk + row) * CD + kof) * 2)); }
        { const int p = tid, row = p >> 2, kof = (p & 3) * 8;
          async_cp16(lds_off(&sV[buf][0]) + (uint32_t)p * 16, vbase,
                     (uint32_t)((row * CSD + kk + kof) * 2)); }
    };

    // Q fragments (16x64 = two K=32 chunks), loaded once
    v16h aq[2];
#pragma unroll
    for (int kc = 0; kc < 2; ++kc)
        aq[kc] = ld_frag(qbase + (size_t)l16 * CD + kc * 32 + ks);

    float m_t = -1e30f, l_t = 0.f;         // running stats for row l16 (both halves)
    v8f cc[4] = {};
    float*    myS = &sS[wave][0];
    _Float16* myP = &sP[wave][0];
    const float scale = 0.125f;            // 1/sqrt(64)

    load_kv(0, 0);
    for (int c = 0; c < CSD / 32; ++c) {
        const int buf = c & 1;
        wait_async0();
        __syncthreads();
        if (c < CSD / 32 - 1) load_kv((c + 1) * 32, buf ^ 1);

        // ---- scores: Q(16x64) @ K^T -> two 16x16 tiles ----
        // hoist all 4 K fragments -> one LDS clause; interleave s0/s1 WMMAs
        v16h kb0 = ld_frag(&sK[buf][l16 * 64 + ks]);
        v16h kb1 = ld_frag(&sK[buf][(16 + l16) * 64 + ks]);
        v16h kb2 = ld_frag(&sK[buf][l16 * 64 + 32 + ks]);
        v16h kb3 = ld_frag(&sK[buf][(16 + l16) * 64 + 32 + ks]);
        v8f s0 = {}, s1 = {};
        s0 = wmma16(aq[0], kb0, s0);
        s1 = wmma16(aq[0], kb1, s1);
        s0 = wmma16(aq[1], kb2, s0);
        s1 = wmma16(aq[1], kb3, s1);

        // scatter scaled scores (C-layout) to wave-private LDS
#pragma unroll
        for (int r = 0; r < 8; ++r) {
            const int row = r + 8 * hig;
            myS[row * 32 + l16]      = s0[r] * scale;
            myS[row * 32 + 16 + l16] = s1[r] * scale;
        }
        // ---- transposed softmax: lane owns row l16, cols hig*16..+15 ----
        float sv[16];
#pragma unroll
        for (int j = 0; j < 4; ++j) {
            v4f tv = *(const v4f*)(myS + l16 * 32 + hig * 16 + j * 4);
#pragma unroll
            for (int e = 0; e < 4; ++e) sv[j * 4 + e] = tv[e];
        }
        float mloc = sv[0];
#pragma unroll
        for (int j = 1; j < 16; ++j) mloc = fmaxf(mloc, sv[j]);
        mloc = fmaxf(mloc, __shfl_xor(mloc, 16, 32));   // single cross-half exchange
        const float mnew  = fmaxf(m_t, mloc);
        const float alpha = __expf(m_t - mnew);
        float lsum = 0.f;
        v8h ph0, ph1;
#pragma unroll
        for (int j = 0; j < 8; ++j) {
            const float p0 = __expf(sv[j] - mnew);
            const float p1 = __expf(sv[j + 8] - mnew);
            lsum += p0 + p1;
            ph0[j] = (_Float16)p0;
            ph1[j] = (_Float16)p1;
        }
        lsum += __shfl_xor(lsum, 16, 32);
        l_t = l_t * alpha + lsum;
        m_t = mnew;
        *(v8h*)(myP + l16 * 32 + hig * 16)     = ph0;
        *(v8h*)(myP + l16 * 32 + hig * 16 + 8) = ph1;
        sA[wave][l16] = alpha;                 // both halves write identical value

        // rescale running context by alpha of each C-layout row
        v4f a0 = *(const v4f*)(&sA[wave][8 * hig]);
        v4f a1 = *(const v4f*)(&sA[wave][8 * hig + 4]);
#pragma unroll
        for (int t = 0; t < 4; ++t)
#pragma unroll
            for (int r = 0; r < 4; ++r) { cc[t][r] *= a0[r]; cc[t][r + 4] *= a1[r]; }

        // P as 16x32 A-fragment
        v8h plo = *(const v8h*)(myP + l16 * 32 + ks);
        v8h phi = *(const v8h*)(myP + l16 * 32 + ks + 16);
        v16h pa = __builtin_shufflevector(plo, phi,
                      0,1,2,3,4,5,6,7,8,9,10,11,12,13,14,15);

        // ---- context accumulate: P(16x32) @ V(32x64) ----
        // hoist all 4 V fragments -> one LDS clause; cc[t] are independent
        v16h vb0 = ld_frag(&sV[buf][(0 * 16 + l16) * 32 + ks]);
        v16h vb1 = ld_frag(&sV[buf][(1 * 16 + l16) * 32 + ks]);
        v16h vb2 = ld_frag(&sV[buf][(2 * 16 + l16) * 32 + ks]);
        v16h vb3 = ld_frag(&sV[buf][(3 * 16 + l16) * 32 + ks]);
        cc[0] = wmma16(pa, vb0, cc[0]);
        cc[1] = wmma16(pa, vb1, cc[1]);
        cc[2] = wmma16(pa, vb2, cc[2]);
        cc[3] = wmma16(pa, vb3, cc[3]);
    }

    // ---- normalize + store ctx (natural f16 layout) ----
    sL[wave][l16] = l_t;
    v4f li0 = *(const v4f*)(&sL[wave][8 * hig]);
    v4f li1 = *(const v4f*)(&sL[wave][8 * hig + 4]);
#pragma unroll
    for (int t = 0; t < 4; ++t)
#pragma unroll
        for (int r = 0; r < 8; ++r) {
            const int row = qrow0 + r + 8 * hig;
            const int col = h * CHD + t * 16 + l16;
            const float dl = (r < 4) ? li0[r] : li1[r - 4];
            ctx[(size_t)row * CD + col] = (_Float16)(cc[t][r] / dl);
        }
}

// ---- launch ----------------------------------------------------------------
extern "C" void kernel_launch(void* const* d_in, const int* in_sizes, int n_in,
                              void* d_out, int out_size, void* d_ws, size_t ws_size,
                              hipStream_t stream) {
    const float* x  = (const float*)d_in[0];
    const float* Wq = (const float*)d_in[1];
    const float* bq = (const float*)d_in[2];
    const float* Wk = (const float*)d_in[3];
    const float* bk = (const float*)d_in[4];
    const float* Wv = (const float*)d_in[5];
    const float* bv = (const float*)d_in[6];
    const float* Wo = (const float*)d_in[7];
    const float* bo = (const float*)d_in[8];
    float* out = (float*)d_out;

    char* ws = (char*)d_ws;
    const size_t NX = (size_t)CB * CS * CD;         // 12,582,912
    const size_t NW = (size_t)CD * CD;              // 589,824
    _Float16* xh   = (_Float16*)(ws);
    _Float16* wqh  = (_Float16*)(ws + 2 * NX);
    _Float16* wkh  = (_Float16*)(ws + 2 * (NX + NW));
    _Float16* wvh  = (_Float16*)(ws + 2 * (NX + 2 * NW));
    _Float16* woh  = (_Float16*)(ws + 2 * (NX + 3 * NW));
    _Float16* qh   = (_Float16*)(ws + 2 * (NX + 4 * NW));
    _Float16* kdh  = (_Float16*)(ws + 2 * (2 * NX + 4 * NW));
    _Float16* vdt  = (_Float16*)(ws + 2 * (2 * NX + 4 * NW) + 2 * (size_t)CB * CSD * CD);
    _Float16* ctxh = (_Float16*)(ws + 2 * (2 * NX + 4 * NW) + 4 * (size_t)CB * CSD * CD);

    dim3 blk(256);

    cast_f32_f16<<<4096, blk, 0, stream>>>(x,  xh,  (int)NX);
    cast_f32_f16<<<1024, blk, 0, stream>>>(Wq, wqh, (int)NW);
    cast_f32_f16<<<1024, blk, 0, stream>>>(Wk, wkh, (int)NW);
    cast_f32_f16<<<1024, blk, 0, stream>>>(Wv, wvh, (int)NW);
    cast_f32_f16<<<1024, blk, 0, stream>>>(Wo, woh, (int)NW);

    // Q: M = 16384 rows; blocks cover 128x128
    gemm_wmma<0><<<dim3(CD / 128, 128), blk, 0, stream>>>(xh, wqh, bq, qh, nullptr);
    // K_d / V_d: M = 2048 gathered rows
    gemm_wmma<1><<<dim3(CD / 128, 16),  blk, 0, stream>>>(xh, wkh, bk, kdh, nullptr);
    gemm_wmma<2><<<dim3(CD / 128, 16),  blk, 0, stream>>>(xh, wvh, bv, vdt, nullptr);

    // attention: B*H*(S/128) = 1536 blocks
    attn_wmma<<<1536, blk, 0, stream>>>(qh, kdh, vdt, ctxh);

    // output projection -> f32 d_out
    gemm_wmma<3><<<dim3(CD / 128, 128), blk, 0, stream>>>(ctxh, woh, bo, nullptr, out);
}
